// Spiking_BasicBlock_87076166959366
// MI455X (gfx1250) — matrix operand order repaired and checked by reading, not compiled, not run on hardware
//
#include <hip/hip_runtime.h>
#include <hip/hip_bf16.h>

// ---------------------------------------------------------------------------
// Spiking BasicBlock on gfx1250: implicit-GEMM conv3x3 via v_wmma_f32_16x16x32_bf16
// (fp32 accumulate), BN folded into scale/bias, sequential IF neuron in VALU.
// Each block: one (bt,h) row -> 32 pixels x 128 output channels.
// 8 waves/block, each wave: 16 co x 32 pixels = 72 WMMAs (A fragment reused 2x).
// Epilogue transposed through LDS for coalesced float4 stores.
// ---------------------------------------------------------------------------

typedef __attribute__((ext_vector_type(16))) __bf16 v16bf;
typedef __attribute__((ext_vector_type(8)))  __bf16 v8bf;
typedef __attribute__((ext_vector_type(8)))  float  v8f;
typedef __attribute__((ext_vector_type(4)))  float  v4f;

#define B_  16
#define T_  7
#define BT_ 112
#define C_  128
#define H_  32
#define W_  32
#define HW_ (H_*W_)
#define BN_EPS 1e-5f
#define INHIB_TH (-0.001f)

// ---------------- pack x: fp32 NCHW -> bf16 [bt][h][w][c] --------------------
__global__ void pack_x_kernel(const float* __restrict__ x, __bf16* __restrict__ xb) {
  int tid = blockIdx.x * blockDim.x + threadIdx.x;       // < BT*H*W*C = 112<<17
  int c  = tid & 127;
  int w  = (tid >> 7)  & 31;
  int h  = (tid >> 12) & 31;
  int bt = tid >> 17;
  xb[tid] = (__bf16)x[(bt*C_ + c)*HW_ + h*W_ + w];
}

// ---------------- pack weights: [O][I][3][3] fp32 -> bf16 [tap][co][ci] ------
__global__ void pack_w_kernel(const float* __restrict__ w, __bf16* __restrict__ wb) {
  int tid = blockIdx.x * blockDim.x + threadIdx.x;       // < 9*128*128 = 9<<14
  int ci  = tid & 127;
  int co  = (tid >> 7) & 127;
  int tap = tid >> 14;
  wb[(tap*C_ + co)*C_ + ci] = (__bf16)w[(co*C_ + ci)*9 + tap];
}

// ---------------- fold BN stats into scale/bias ------------------------------
__global__ void bn_prep_kernel(const float* __restrict__ g1, const float* __restrict__ b1,
                               const float* __restrict__ m1, const float* __restrict__ v1,
                               const float* __restrict__ g2, const float* __restrict__ b2,
                               const float* __restrict__ m2, const float* __restrict__ v2,
                               float* __restrict__ bn) {   // [2][2][128]: scale,bias per BN
  int i = threadIdx.x & 127;
  int which = threadIdx.x >> 7;                            // 0 or 1
  const float* g = which ? g2 : g1;
  const float* b = which ? b2 : b1;
  const float* m = which ? m2 : m1;
  const float* v = which ? v2 : v1;
  float s = g[i] * rsqrtf(v[i] + BN_EPS);
  bn[which*256 + i]       = s;
  bn[which*256 + 128 + i] = b[i] - m[i]*s;
}

// ---------------- implicit-GEMM conv3x3 + BN (+identity) via WMMA ------------
// inb : bf16 [BT][H][W][C] (channels-last), wb: bf16 [9][co][ci]
// ident: fp32 NCHW (reshaped BT,C,H,W) or nullptr; yout: fp32 [BT][H][W][C]
__global__ __launch_bounds__(256)
void conv_wmma_kernel(const __bf16* __restrict__ inb,
                      const __bf16* __restrict__ wb,
                      const float* __restrict__ scale,
                      const float* __restrict__ bias,
                      const float* __restrict__ ident,
                      float* __restrict__ yout)
{
  __shared__ alignas(32) __bf16 patch[3][34][C_];          // 26,112 B halo row
  __shared__ alignas(16) float  trbuf[8][32][17];          // padded transpose buffer

  int blk = blockIdx.x;                                    // BT*H = 3584 blocks
  int h   = blk & 31;
  int bt  = blk >> 5;
  int tid = threadIdx.x;

  // Stage 3x34x128 bf16 halo row into LDS (zero-padded borders), 16B chunks,
  // consecutive threads -> consecutive 16B (fully coalesced global_load_b128).
  for (int c = tid; c < 3*34*16; c += 256) {
    int row = c / (34*16);
    int rem = c - row*(34*16);
    int col = rem >> 4;
    int ci8 = rem & 15;
    int hin = h - 1 + row;
    int win = col - 1;
    v8bf v = {};
    if (hin >= 0 && hin < H_ && (unsigned)win < (unsigned)W_)
      v = *(const v8bf*)(inb + ((bt*H_ + hin)*W_ + win)*C_ + ci8*8);
    *(v8bf*)(&patch[row][col][ci8*8]) = v;
  }
  __syncthreads();

  int lane   = tid & 31;
  int waveId = tid >> 5;
  int co0    = waveId * 16;                                // 8 waves x 16 co = 128
  int n      = lane & 15;                                  // pixel column in tile
  int half   = lane >> 4;
  int kbA    = half * 8;                                   // unused; A frag uses kbB like B
  int kbB    = half * 16;                                  // B-frag K base

  __builtin_prefetch(wb + (size_t)co0 * C_, 0, 3);         // global_prefetch_b8

  v8f acc0 = {}, acc1 = {};                                // pixels [0,16) and [16,32)
  #pragma unroll
  for (int tap = 0; tap < 9; ++tap) {
    int kh = tap / 3, kw = tap - kh*3;
    const __bf16* wtap = wb + ((tap*C_ + (co0 + n)) * C_); // A row m = lane&15 -> co0+m
    const __bf16* bsrc = &patch[kh][n + kw][0];
    #pragma unroll
    for (int kb = 0; kb < 4; ++kb) {                       // K = 128 ci in 4 x 32
      int k0 = kb * 32;
      // A fragment: lane half selects K groups {0..15} / {16..31}
      v8bf alo = *(const v8bf*)(wtap + k0 + kbB);
      v8bf ahi = *(const v8bf*)(wtap + k0 + kbB + 8);
      v16bf afrag;
      #pragma unroll
      for (int i = 0; i < 8; ++i) { afrag[i] = alo[i]; afrag[i+8] = ahi[i]; }
      // B fragments: 16 contiguous bf16 along ci (32 B) from LDS; A reused 2x.
      v16bf bf0 = *(const v16bf*)(bsrc + k0 + kbB);
      v16bf bf1 = *(const v16bf*)(bsrc + 16*C_ + k0 + kbB);
      acc0 = __builtin_amdgcn_wmma_f32_16x16x32_bf16(
          false, afrag, false, bf0, (short)0, acc0, false, false);
      acc1 = __builtin_amdgcn_wmma_f32_16x16x32_bf16(
          false, afrag, false, bf1, (short)0, acc1, false, false);
    }
  }

  // Epilogue: D layout -> VGPR r: lanes 0-15 M=r, lanes 16-31 M=r+8; N = lane&15.
  // BN + identity in-register (identity reads are coalesced: fixed co, lanes
  // sweep w), then transpose via LDS for coalesced float4 global stores.
  #pragma unroll
  for (int r = 0; r < 8; ++r) {
    int m  = r + half*8;                                   // co within wave tile
    int co = co0 + m;
    float y0 = acc0[r] * scale[co] + bias[co];
    float y1 = acc1[r] * scale[co] + bias[co];
    if (ident) {
      const float* ip = ident + (bt*C_ + co)*HW_ + h*W_;
      y0 += ip[n];
      y1 += ip[16 + n];
    }
    trbuf[waveId][n][m]      = y0;
    trbuf[waveId][16 + n][m] = y1;
  }
  __syncthreads();

  // Lane j owns pixel w=j of this row: 16 contiguous co values -> 4x float4.
  float* yp = yout + ((bt*H_ + h)*W_ + lane)*C_ + co0;
  const float* tr = &trbuf[waveId][lane][0];
  #pragma unroll
  for (int q = 0; q < 4; ++q) {
    v4f v = { tr[4*q+0], tr[4*q+1], tr[4*q+2], tr[4*q+3] };
    *(v4f*)(yp + 4*q) = v;
  }
}

// ---------------- IF neuron scan (T sequential), stage 1 ---------------------
// y: fp32 [BT][H][W][C] -> out1: bf16 [BT][H][W][C], value = (spike-inhibit)*th
__global__ void if1_kernel(const float* __restrict__ y, const float* __restrict__ alpha,
                           __bf16* __restrict__ outb) {
  int tid = blockIdx.x * blockDim.x + threadIdx.x;         // < B*H*W*C = 16<<17
  int c = tid & 127;
  int w = (tid >> 7)  & 31;
  int h = (tid >> 12) & 31;
  int b = tid >> 17;
  float th = alpha[0];
  float mem = 0.5f * th;
  float ssum = 0.f;
  int base = (b*T_*H_ + h)*W_ + w;                         // pixel index at t=0
  #pragma unroll
  for (int t = 0; t < T_; ++t) {
    int pix = base + t*HW_;
    mem += y[pix*C_ + c];
    float sp = (mem >= th) ? 1.f : 0.f;
    mem -= th * sp;
    ssum += sp;
    float inh = ((mem <= INHIB_TH) && (ssum > 0.f)) ? 1.f : 0.f;
    mem += th * inh;
    ssum -= inh;
    outb[pix*C_ + c] = (__bf16)((sp - inh) * th);
  }
}

// ---------------- IF neuron scan, stage 2 -> final fp32 NCHW output ----------
__global__ void if2_kernel(const float* __restrict__ y, const float* __restrict__ alpha,
                           float* __restrict__ out) {
  int tid = blockIdx.x * blockDim.x + threadIdx.x;
  int c = tid & 127;
  int w = (tid >> 7)  & 31;
  int h = (tid >> 12) & 31;
  int b = tid >> 17;
  float th = alpha[0];
  float mem = 0.5f * th;
  float ssum = 0.f;
  int base = (b*T_*H_ + h)*W_ + w;
  #pragma unroll
  for (int t = 0; t < T_; ++t) {
    int pix = base + t*HW_;
    mem += y[pix*C_ + c];
    float sp = (mem >= th) ? 1.f : 0.f;
    mem -= th * sp;
    ssum += sp;
    float inh = ((mem <= INHIB_TH) && (ssum > 0.f)) ? 1.f : 0.f;
    mem += th * inh;
    ssum -= inh;
    out[(((b*T_ + t)*C_ + c)*H_ + h)*W_ + w] = (sp - inh) * th;
  }
}

// ---------------------------------------------------------------------------
extern "C" void kernel_launch(void* const* d_in, const int* in_sizes, int n_in,
                              void* d_out, int out_size, void* d_ws, size_t ws_size,
                              hipStream_t stream) {
  const float* x   = (const float*)d_in[0];
  const float* w1  = (const float*)d_in[1];
  const float* g1  = (const float*)d_in[2];
  const float* be1 = (const float*)d_in[3];
  const float* m1  = (const float*)d_in[4];
  const float* va1 = (const float*)d_in[5];
  const float* a1  = (const float*)d_in[6];
  const float* w2  = (const float*)d_in[7];
  const float* g2  = (const float*)d_in[8];
  const float* be2 = (const float*)d_in[9];
  const float* m2  = (const float*)d_in[10];
  const float* va2 = (const float*)d_in[11];
  const float* a2  = (const float*)d_in[12];
  float* out = (float*)d_out;

  // Workspace layout (needs ~118.1 MB):
  //   [0,            2N)  xb    : bf16 x, channels-last
  //   [2N,           4N)  out1b : bf16 spike output of stage 1
  //   [4N,           8N)  ybuf  : fp32 pre-activation (reused y1 then y2; safe,
  //                               stream-ordered: if1 drains y1 before conv2 writes)
  //   [8N, 8N+2*294912)   wb1, wb2 : bf16 packed weights [tap][co][ci]
  //   then 512 floats     bn    : {scale1,bias1,scale2,bias2}
  const size_t NEL = (size_t)BT_ * C_ * HW_;               // 14,680,064
  char* ws = (char*)d_ws;
  __bf16* xb    = (__bf16*)(ws);
  __bf16* out1b = (__bf16*)(ws + NEL*2);
  float*  ybuf  = (float*) (ws + NEL*4);
  __bf16* wb1   = (__bf16*)(ws + NEL*8);
  __bf16* wb2   = wb1 + 9*C_*C_;
  float*  bn    = (float*) (ws + NEL*8 + (size_t)2*9*C_*C_*2);

  pack_x_kernel<<<(int)(NEL/256), 256, 0, stream>>>(x, xb);
  pack_w_kernel<<<(9*C_*C_)/256, 256, 0, stream>>>(w1, wb1);
  pack_w_kernel<<<(9*C_*C_)/256, 256, 0, stream>>>(w2, wb2);
  bn_prep_kernel<<<1, 256, 0, stream>>>(g1, be1, m1, va1, g2, be2, m2, va2, bn);

  // conv1 + BN1 -> y1
  conv_wmma_kernel<<<BT_*H_, 256, 0, stream>>>(xb, wb1, bn, bn+128, nullptr, ybuf);
  // IF over T -> out1 (bf16, channels-last)
  if1_kernel<<<(B_*HW_*C_)/256, 256, 0, stream>>>(ybuf, a1, out1b);
  // conv2 + BN2 + identity(x) -> y2 (reuse ybuf)
  conv_wmma_kernel<<<BT_*H_, 256, 0, stream>>>(out1b, wb2, bn+256, bn+384, x, ybuf);
  // IF over T -> final fp32 NCHW output
  if2_kernel<<<(B_*HW_*C_)/256, 256, 0, stream>>>(ybuf, a2, out);
}
